// MultiTaskGNNSurrogate_43593918054659
// MI455X (gfx1250) — matrix operand order repaired and checked by reading, not compiled, not run on hardware
//
#include <hip/hip_runtime.h>

// ---------------------------------------------------------------------------
// MI455X (gfx1250) implementation.
// Dense matmuls -> v_wmma_f32_16x16x32_f16 (f16 operands staged via LDS,
// fp32 accumulate), 64x64 block tile, wave computes 16x32 (2 WMMA/K-step),
// compile-time K and accumulate-flag so the hot loop and epilogue are
// straight-line. Irregular graph scatter/gather, LN, LSTM pointwise and the
// tiny 6x6 attention stay on VALU/DS paths.
// ---------------------------------------------------------------------------

typedef __attribute__((ext_vector_type(16))) _Float16 v16h;
typedef __attribute__((ext_vector_type(8)))  float    v8f;
typedef __attribute__((ext_vector_type(8)))  _Float16 h8v;
typedef __attribute__((ext_vector_type(4)))  _Float16 h4v;
typedef __attribute__((ext_vector_type(4)))  float    f4v;

#define T_STEPS 6
#define HDIM    128
#define RDIM    128
#define EDIM    16
#define FIN     32

__device__ __forceinline__ float softplus_f(float x) {
  return (x > 20.0f) ? x : log1pf(__expf(x));
}
__device__ __forceinline__ float sigmoid_f(float x) {
  return 1.0f / (1.0f + __expf(-x));
}

// ---------------------------------------------------------------------------
// zero fill
// ---------------------------------------------------------------------------
__global__ void zero_kernel(float* __restrict__ p, long n) {
  long i = (long)blockIdx.x * blockDim.x + threadIdx.x;
  if (i < n) p[i] = 0.0f;
}

// ---------------------------------------------------------------------------
// Per-edge, per-layer message coefficients:
//   sgain = sign * gain, sbias = sign * bias   (sign = 2*dir - 1)
// ---------------------------------------------------------------------------
__global__ void edge_coeff_kernel(const float* __restrict__ edge_attr,
                                  const int* __restrict__ edge_type,
                                  const float* __restrict__ ete,
                                  const float* __restrict__ w0, const float* __restrict__ b0,
                                  const float* __restrict__ w1, const float* __restrict__ b1,
                                  const float* __restrict__ w2, const float* __restrict__ b2,
                                  float* __restrict__ sgain, float* __restrict__ sbias,
                                  int E) {
  int e = blockIdx.x * blockDim.x + threadIdx.x;
  if (e >= E) return;
  const float* ea = edge_attr + (long)e * EDIM;
  float dir = ea[EDIM - 2];
  float spd = ea[EDIM - 1];
  int t = edge_type[e];
  const float* et = ete + t * EDIM;
  float eav[EDIM];
#pragma unroll
  for (int j = 0; j < EDIM; ++j) eav[j] = ea[j] + et[j];

  float gain = 0.0f, biasv = 0.0f;
  if (t == 1) {  // PUMP: 2-row MLP, scaled by pump speed
    float r0 = b1[0], r1 = b1[1];
#pragma unroll
    for (int j = 0; j < EDIM; ++j) {
      r0 += w1[j] * eav[j];
      r1 += w1[EDIM + j] * eav[j];
    }
    float scale = (dir > 0.0f) ? dir : 1.0f;
    float speed = spd * scale;
    gain = softplus_f(r0) * speed;
    biasv = r1 * speed;
  } else {
    const float* w = (t == 0) ? w0 : w2;
    const float* b = (t == 0) ? b0 : b2;
    float r0 = b[0];
#pragma unroll
    for (int j = 0; j < EDIM; ++j) r0 += w[j] * eav[j];
    gain = softplus_f(r0);
  }
  float sign = dir * 2.0f - 1.0f;
  sgain[e] = sign * gain;
  sbias[e] = sign * biasv;
}

// ---------------------------------------------------------------------------
// Scatter: aggr[dst] += sgain*(x[src]-x[dst]) + sbias   (per feature)
// ---------------------------------------------------------------------------
__global__ void aggregate_kernel(const float* __restrict__ x,
                                 const float* __restrict__ sg,
                                 const float* __restrict__ sb,
                                 const int* __restrict__ src,
                                 const int* __restrict__ dst,
                                 float* __restrict__ aggr,
                                 int E, int C) {
  long idx = (long)blockIdx.x * blockDim.x + threadIdx.x;
  if (idx >= (long)E * C) return;
  int e = (int)(idx / C);
  int c = (int)(idx % C);
  int s = src[e], d = dst[e];
  float msg = sg[e] * (x[(long)s * C + c] - x[(long)d * C + c]) + sb[e];
  atomicAdd(&aggr[(long)d * C + c], msg);
}

// ---------------------------------------------------------------------------
// WMMA GEMM: C[M,N] (+)= A[M,K] @ W[N,K]^T + bias[N]
// 256 threads = 8 waves; block tile 64x64; each wave owns a 16x32 strip:
// one A fragment x two B fragments -> 2 v_wmma per K-step. K and ACC are
// template parameters (K in {32,128} here) -> fully unrolled straight-line
// hot loop and branch-free epilogue.
// CONTRACT: N must be a multiple of GBN (true for all call sites: 128/384/512).
// Out-of-range M rows are CLAMPED (not masked): in D = A.B a bogus row only
// pollutes its own output row, which the epilogue discards. Keeps staging
// branch-free -> clean global_load_b128 + ds_store_b64.
// ---------------------------------------------------------------------------
#define GBM 64
#define GBN 64
#define GBK 32

template <int K, int ACC>
__global__ __launch_bounds__(256)
void wmma_gemm_kernel(const float* __restrict__ A, const float* __restrict__ W,
                      const float* __restrict__ bias, float* __restrict__ C,
                      int M, int N) {
  __shared__ _Float16 As[GBM][GBK + 8];   // row stride 40 halves = 80 B
  __shared__ _Float16 Ws[GBN][GBK + 8];

  const int tid  = threadIdx.x;
  const int lane = tid & 31;
  const int wave = tid >> 5;      // 0..7
  const int wm   = wave & 3;      // 0..3  -> 16-row sub-tile
  const int wn   = wave >> 2;     // 0..1  -> 32-col strip
  const int bm   = blockIdx.x * GBM;
  const int bn   = blockIdx.y * GBN;

  v8f acc0 = {0.f, 0.f, 0.f, 0.f, 0.f, 0.f, 0.f, 0.f};
  v8f acc1 = {0.f, 0.f, 0.f, 0.f, 0.f, 0.f, 0.f, 0.f};

#pragma unroll
  for (int k0 = 0; k0 < K; k0 += GBK) {
    // ---- stage A tile: 64 rows x 8 float4 chunks = 512; 2 per thread ----
#pragma unroll
    for (int i = 0; i < 2; ++i) {
      int idx = tid + i * 256;          // 0..511
      int r   = idx >> 3;               // 0..63
      int c4  = idx & 7;                // float4 chunk
      int gr  = bm + r;
      int crow = (gr < M) ? gr : (M - 1);   // clamp, no branch
      f4v v = *((const f4v*)(A + (long)crow * K + k0) + c4);
      if (k0 + GBK < K)  // prefetch next K tile (global_prefetch_b8)
        __builtin_prefetch(A + (long)crow * K + k0 + GBK + c4 * 4, 0, 1);
      h4v h;
      h[0] = (_Float16)v[0]; h[1] = (_Float16)v[1];
      h[2] = (_Float16)v[2]; h[3] = (_Float16)v[3];
      *(h4v*)(&As[r][c4 * 4]) = h;
    }
    // ---- stage W tile: 64 rows x 8 chunks = 512; 2 per thread ----
#pragma unroll
    for (int i = 0; i < 2; ++i) {
      int idx = tid + i * 256;
      int r   = idx >> 3;
      int c4  = idx & 7;
      int gn  = bn + r;    // always < N (N % GBN == 0 by contract)
      f4v v = *((const f4v*)(W + (long)gn * K + k0) + c4);
      h4v h;
      h[0] = (_Float16)v[0]; h[1] = (_Float16)v[1];
      h[2] = (_Float16)v[2]; h[3] = (_Float16)v[3];
      *(h4v*)(&Ws[r][c4 * 4]) = h;
    }
    __syncthreads();

    // ---- fragments (contiguous k = 16*(lane>>4) + s, same order in A and B) ----
    const int arow = wm * 16 + (lane & 15);
    const int hiA  = (lane >> 4) << 4;   // 0 or 16
    h8v a0 = *(const h8v*)(&As[arow][hiA]);
    h8v a1 = *(const h8v*)(&As[arow][hiA + 8]);
    v16h afrag;
#pragma unroll
    for (int h = 0; h < 8; ++h) { afrag[h] = a0[h]; afrag[8 + h] = a1[h]; }

    const int bcol = wn * 32 + (lane & 15);
    const int hiB  = (lane >> 4) << 4;   // 0 or 16
    h8v b00 = *(const h8v*)(&Ws[bcol][hiB]);
    h8v b01 = *(const h8v*)(&Ws[bcol][hiB + 8]);
    h8v b10 = *(const h8v*)(&Ws[bcol + 16][hiB]);
    h8v b11 = *(const h8v*)(&Ws[bcol + 16][hiB + 8]);
    v16h bfrag0, bfrag1;
#pragma unroll
    for (int h = 0; h < 8; ++h) {
      bfrag0[h] = b00[h]; bfrag0[8 + h] = b01[h];
      bfrag1[h] = b10[h]; bfrag1[8 + h] = b11[h];
    }

    acc0 = __builtin_amdgcn_wmma_f32_16x16x32_f16(false, afrag, false, bfrag0,
                                                  (short)0, acc0, false, false);
    acc1 = __builtin_amdgcn_wmma_f32_16x16x32_f16(false, afrag, false, bfrag1,
                                                  (short)0, acc1, false, false);
    __syncthreads();
  }

  // ---- epilogue: D layout VGPR r -> M = r + 8*(lane>=16); N = lane&15 ----
  const int col0  = bn + wn * 32 + (lane & 15);   // always < N by contract
  const int col1  = col0 + 16;
  const int rbase = bm + wm * 16 + ((lane >> 4) << 3);
  const float bv0 = bias[col0];
  const float bv1 = bias[col1];
#pragma unroll
  for (int r = 0; r < 8; ++r) {
    int grow = rbase + r;
    if (grow < M) {
      long o0 = (long)grow * N + col0;
      long o1 = (long)grow * N + col1;
      float v0 = acc0[r] + bv0;
      float v1 = acc1[r] + bv1;
      if (ACC) { v0 += C[o0]; v1 += C[o1]; }
      C[o0] = v0;
      C[o1] = v1;
    }
  }
}

// ---------------------------------------------------------------------------
// Per-node: select type output, ReLU, LayerNorm, optional residual.
// One block (128 threads) per node.
// ---------------------------------------------------------------------------
__global__ __launch_bounds__(128)
void select_ln_kernel(const float* __restrict__ y0, const float* __restrict__ y1,
                      const int* __restrict__ node_type,
                      const float* __restrict__ g, const float* __restrict__ b,
                      const float* __restrict__ identity,
                      float* __restrict__ out, int Nn) {
  __shared__ float red[4];
  int n = blockIdx.x;
  int f = threadIdx.x;
  const float* y = (node_type[n] == 0) ? y0 : y1;
  float v = y[(long)n * HDIM + f];
  v = fmaxf(v, 0.0f);

  float s = v;
#pragma unroll
  for (int o = 16; o > 0; o >>= 1) s += __shfl_xor(s, o, 32);
  if ((f & 31) == 0) red[f >> 5] = s;
  __syncthreads();
  float mu = (red[0] + red[1] + red[2] + red[3]) * (1.0f / HDIM);
  __syncthreads();

  float d = v - mu;
  float q = d * d;
#pragma unroll
  for (int o = 16; o > 0; o >>= 1) q += __shfl_xor(q, o, 32);
  if ((f & 31) == 0) red[f >> 5] = q;
  __syncthreads();
  float var = (red[0] + red[1] + red[2] + red[3]) * (1.0f / HDIM);

  float r = rsqrtf(var + 1e-5f);
  float o2 = g[f] * d * r + b[f];
  if (identity) o2 += identity[(long)n * HDIM + f];
  out[(long)n * HDIM + f] = o2;
}

// ---------------------------------------------------------------------------
// LSTM pointwise: gates (N,4R) in [i|f|g|o] order; update c,h; hs[n,t,:] = h
// ---------------------------------------------------------------------------
__global__ void lstm_pointwise_kernel(const float* __restrict__ gates,
                                      float* __restrict__ c, float* __restrict__ h,
                                      float* __restrict__ hs, int Nn, int t) {
  long idx = (long)blockIdx.x * blockDim.x + threadIdx.x;
  if (idx >= (long)Nn * RDIM) return;
  int n = (int)(idx >> 7);
  int r = (int)(idx & 127);
  const float* gr = gates + (long)n * (4 * RDIM);
  float gi = sigmoid_f(gr[r]);
  float gf = sigmoid_f(gr[RDIM + r]);
  float gg = tanhf(gr[2 * RDIM + r]);
  float go = sigmoid_f(gr[3 * RDIM + r]);
  float cn = gf * c[idx] + gi * gg;
  float hn = go * tanhf(cn);
  c[idx] = cn;
  h[idx] = hn;
  hs[((long)n * T_STEPS + t) * RDIM + r] = hn;
}

// ---------------------------------------------------------------------------
// MHA over the T=6 sequence, one wave per (node, head); lane = d-index (d=32)
// ---------------------------------------------------------------------------
__global__ __launch_bounds__(256)
void attention_kernel(const float* __restrict__ qkv, float* __restrict__ ctx, int Nn) {
  long wid = ((long)blockIdx.x * blockDim.x + threadIdx.x) >> 5;
  int lane = threadIdx.x & 31;
  if (wid >= (long)Nn * 4) return;
  int n = (int)(wid >> 2);
  int hh = (int)(wid & 3);

  float q[T_STEPS], k[T_STEPS], v[T_STEPS];
#pragma unroll
  for (int t = 0; t < T_STEPS; ++t) {
    const float* row = qkv + ((long)n * T_STEPS + t) * (3 * RDIM) + hh * 32 + lane;
    q[t] = row[0];
    k[t] = row[RDIM];
    v[t] = row[2 * RDIM];
  }
  const float scale = 0.17677669529663687f;  // 1/sqrt(32)
#pragma unroll
  for (int i = 0; i < T_STEPS; ++i) {
    float s[T_STEPS];
    float mx = -1e30f;
#pragma unroll
    for (int j = 0; j < T_STEPS; ++j) {
      float p = q[i] * k[j];
#pragma unroll
      for (int o = 16; o > 0; o >>= 1) p += __shfl_xor(p, o, 32);
      p *= scale;
      s[j] = p;
      mx = fmaxf(mx, p);
    }
    float denom = 0.0f;
#pragma unroll
    for (int j = 0; j < T_STEPS; ++j) { s[j] = __expf(s[j] - mx); denom += s[j]; }
    float inv = 1.0f / denom;
    float a = 0.0f;
#pragma unroll
    for (int j = 0; j < T_STEPS; ++j) a += s[j] * v[j];
    ctx[((long)n * T_STEPS + i) * RDIM + hh * 32 + lane] = a * inv;
  }
}

// ---------------------------------------------------------------------------
// Node head: out[n,t,0..1] = h_final[n,t] . nd_w + nd_b (+ X_seq[t,n,1] on ch1)
// ---------------------------------------------------------------------------
__global__ __launch_bounds__(256)
void node_head_kernel(const float* __restrict__ hf, const float* __restrict__ nd_w,
                      const float* __restrict__ nd_b, const float* __restrict__ X_seq,
                      float* __restrict__ out, int Nn) {
  __shared__ float w[2 * RDIM];
  if (threadIdx.x < 2 * RDIM) w[threadIdx.x] = nd_w[threadIdx.x];
  __syncthreads();
  long idx = (long)blockIdx.x * blockDim.x + threadIdx.x;
  if (idx >= (long)Nn * T_STEPS) return;
  int n = (int)(idx / T_STEPS), t = (int)(idx % T_STEPS);
  const float* row = hf + idx * RDIM;
  float d0 = 0.0f, d1 = 0.0f;
#pragma unroll 4
  for (int f = 0; f < RDIM; ++f) {
    float hv = row[f];
    d0 += w[f] * hv;
    d1 += w[RDIM + f] * hv;
  }
  out[idx * 2 + 0] = d0 + nd_b[0];
  out[idx * 2 + 1] = d1 + nd_b[1] + X_seq[((long)t * Nn + n) * FIN + 1];
}

// wA = ed_w[0:128]+ed_w[256:384], wB = ed_w[128:256]-ed_w[256:384]
__global__ void edge_prep_kernel(const float* __restrict__ ed_w, float* __restrict__ wAB) {
  int f = threadIdx.x;
  wAB[f]        = ed_w[f] + ed_w[2 * RDIM + f];
  wAB[RDIM + f] = ed_w[RDIM + f] - ed_w[2 * RDIM + f];
}

// Edge head: one wave per (e,t); lane-parallel dot + shuffle reduce
__global__ __launch_bounds__(256)
void edge_head_kernel(const float* __restrict__ hf, const float* __restrict__ wAB,
                      const float* __restrict__ ed_b, const int* __restrict__ src,
                      const int* __restrict__ dst, float* __restrict__ out, long total) {
  __shared__ float w[2 * RDIM];
  if (threadIdx.x < 2 * RDIM) w[threadIdx.x] = wAB[threadIdx.x];
  __syncthreads();
  long wid = ((long)blockIdx.x * blockDim.x + threadIdx.x) >> 5;
  int lane = threadIdx.x & 31;
  if (wid >= total) return;
  long e = wid / T_STEPS;
  int t = (int)(wid % T_STEPS);
  const float* hsrc = hf + ((long)src[e] * T_STEPS + t) * RDIM;
  const float* hdst = hf + ((long)dst[e] * T_STEPS + t) * RDIM;
  float p = 0.0f;
#pragma unroll
  for (int i = 0; i < 4; ++i) {
    int f = lane + i * 32;
    p += w[f] * hsrc[f] + w[RDIM + f] * hdst[f];
  }
#pragma unroll
  for (int o = 16; o > 0; o >>= 1) p += __shfl_xor(p, o, 32);
  if (lane == 0) out[wid] = p + ed_b[0];
}

// ---------------------------------------------------------------------------
// Host orchestration
// ---------------------------------------------------------------------------
extern "C" void kernel_launch(void* const* d_in, const int* in_sizes, int n_in,
                              void* d_out, int out_size, void* d_ws, size_t ws_size,
                              hipStream_t stream) {
  // --- input pointers (JAX pytree flatten order: dict keys sorted) ---
  const float* X_seq     = (const float*)d_in[0];
  const float* edge_attr = (const float*)d_in[1];
  const float* ed_b      = (const float*)d_in[2];
  const float* ed_w      = (const float*)d_in[3];
  const float* fc_b      = (const float*)d_in[4];
  const float* fc_w      = (const float*)d_in[5];
  struct LayerP {
    const float *eb0, *eb1, *eb2, *ew0, *ew1, *ew2, *ete, *lin_b, *lin_w, *ln_b, *ln_g;
  } L[2];
  for (int li = 0; li < 2; ++li) {
    int base = 6 + li * 11;
    L[li].eb0   = (const float*)d_in[base + 0];
    L[li].eb1   = (const float*)d_in[base + 1];
    L[li].eb2   = (const float*)d_in[base + 2];
    L[li].ew0   = (const float*)d_in[base + 3];
    L[li].ew1   = (const float*)d_in[base + 4];
    L[li].ew2   = (const float*)d_in[base + 5];
    L[li].ete   = (const float*)d_in[base + 6];
    L[li].lin_b = (const float*)d_in[base + 7];
    L[li].lin_w = (const float*)d_in[base + 8];
    L[li].ln_b  = (const float*)d_in[base + 9];
    L[li].ln_g  = (const float*)d_in[base + 10];
  }
  const float* bhh   = (const float*)d_in[28];
  const float* bih   = (const float*)d_in[29];
  const float* whh   = (const float*)d_in[30];
  const float* wih   = (const float*)d_in[31];
  const float* in_b  = (const float*)d_in[32];
  const float* in_w  = (const float*)d_in[33];
  const float* out_b = (const float*)d_in[34];
  const float* out_w = (const float*)d_in[35];
  const float* nd_b  = (const float*)d_in[36];
  const float* nd_w  = (const float*)d_in[37];
  const int*   eidx  = (const int*)d_in[38];
  const int*   ntype = (const int*)d_in[39];
  const int*   etype = (const int*)d_in[40];

  const int Nn = in_sizes[39];
  const int E  = in_sizes[40];
  const int* src = eidx;
  const int* dst = eidx + E;

  // --- workspace layout ---
  float* ws = (float*)d_ws;
  long off = 0;
  auto alloc = [&](long n) { float* p = ws + off; off += (n + 31) & ~31L; return p; };
  float* sg0    = alloc(E);
  float* sb0    = alloc(E);
  float* sg1    = alloc(E);
  float* sb1    = alloc(E);
  float* aggr   = alloc((long)Nn * HDIM);
  float* y0     = alloc((long)Nn * HDIM);
  float* y1     = alloc((long)Nn * HDIM);
  float* henc0  = alloc((long)Nn * HDIM);
  float* henc1  = alloc((long)Nn * HDIM);
  float* emb    = alloc((long)T_STEPS * Nn * HDIM);
  float* gates  = alloc((long)Nn * 4 * RDIM);
  float* lstm_h = alloc((long)Nn * RDIM);
  float* lstm_c = alloc((long)Nn * RDIM);
  float* hs     = alloc((long)Nn * T_STEPS * RDIM);
  float* qkv    = alloc((long)Nn * T_STEPS * 3 * RDIM);
  float* ctx    = alloc((long)Nn * T_STEPS * RDIM);
  float* hfinal = alloc((long)Nn * T_STEPS * RDIM);
  float* wAB    = alloc(2 * RDIM);

  auto cdiv = [](long a, long b) { return (a + b - 1) / b; };
  auto zero = [&](float* p, long n) {
    zero_kernel<<<dim3((unsigned)cdiv(n, 256)), 256, 0, stream>>>(p, n);
  };
  // N must be a multiple of 64 (true for 128/384/512); K in {32,128}.
  auto gemm = [&](const float* A, const float* W, const float* bias, float* C,
                  int M, int Ncol, int K, int acc) {
    dim3 g((unsigned)cdiv(M, GBM), (unsigned)(Ncol / GBN));
    if (K == 32)
      wmma_gemm_kernel<32, 0><<<g, 256, 0, stream>>>(A, W, bias, C, M, Ncol);
    else if (acc)
      wmma_gemm_kernel<128, 1><<<g, 256, 0, stream>>>(A, W, bias, C, M, Ncol);
    else
      wmma_gemm_kernel<128, 0><<<g, 256, 0, stream>>>(A, W, bias, C, M, Ncol);
  };

  // --- per-layer edge coefficients (time-invariant) ---
  edge_coeff_kernel<<<dim3((unsigned)cdiv(E, 256)), 256, 0, stream>>>(
      edge_attr, etype, L[0].ete, L[0].ew0, L[0].eb0, L[0].ew1, L[0].eb1,
      L[0].ew2, L[0].eb2, sg0, sb0, E);
  edge_coeff_kernel<<<dim3((unsigned)cdiv(E, 256)), 256, 0, stream>>>(
      edge_attr, etype, L[1].ete, L[1].ew0, L[1].eb0, L[1].ew1, L[1].eb1,
      L[1].ew2, L[1].eb2, sg1, sb1, E);

  // --- encoder, per time step ---
  for (int t = 0; t < T_STEPS; ++t) {
    const float* x0 = X_seq + (long)t * Nn * FIN;
    // layer 0 (cin = 32, no residual)
    zero(aggr, (long)Nn * FIN);
    aggregate_kernel<<<dim3((unsigned)cdiv((long)E * FIN, 256)), 256, 0, stream>>>(
        x0, sg0, sb0, src, dst, aggr, E, FIN);
    gemm(aggr, L[0].lin_w, L[0].lin_b, y0, Nn, HDIM, FIN, 0);
    gemm(aggr, L[0].lin_w + (long)HDIM * FIN, L[0].lin_b + HDIM, y1, Nn, HDIM, FIN, 0);
    select_ln_kernel<<<dim3((unsigned)Nn), 128, 0, stream>>>(
        y0, y1, ntype, L[0].ln_g, L[0].ln_b, nullptr, henc0, Nn);
    // layer 1 (cin = 128, residual)
    zero(aggr, (long)Nn * HDIM);
    aggregate_kernel<<<dim3((unsigned)cdiv((long)E * HDIM, 256)), 256, 0, stream>>>(
        henc0, sg1, sb1, src, dst, aggr, E, HDIM);
    gemm(aggr, L[1].lin_w, L[1].lin_b, y0, Nn, HDIM, HDIM, 0);
    gemm(aggr, L[1].lin_w + (long)HDIM * HDIM, L[1].lin_b + HDIM, y1, Nn, HDIM, HDIM, 0);
    select_ln_kernel<<<dim3((unsigned)Nn), 128, 0, stream>>>(
        y0, y1, ntype, L[1].ln_g, L[1].ln_b, henc0, henc1, Nn);
    // fc -> emb[t]
    gemm(henc1, fc_w, fc_b, emb + (long)t * Nn * HDIM, Nn, HDIM, HDIM, 0);
  }

  // --- LSTM over T ---
  zero(lstm_h, (long)Nn * RDIM);
  zero(lstm_c, (long)Nn * RDIM);
  for (int t = 0; t < T_STEPS; ++t) {
    gemm(emb + (long)t * Nn * HDIM, wih, bih, gates, Nn, 4 * RDIM, HDIM, 0);
    gemm(lstm_h, whh, bhh, gates, Nn, 4 * RDIM, RDIM, 1);
    lstm_pointwise_kernel<<<dim3((unsigned)cdiv((long)Nn * RDIM, 256)), 256, 0, stream>>>(
        gates, lstm_c, lstm_h, hs, Nn, t);
  }

  // --- MHA ---
  gemm(hs, in_w, in_b, qkv, Nn * T_STEPS, 3 * RDIM, RDIM, 0);
  attention_kernel<<<dim3((unsigned)cdiv((long)Nn * 4, 8)), 256, 0, stream>>>(qkv, ctx, Nn);
  hipMemcpyAsync(hfinal, hs, (size_t)Nn * T_STEPS * RDIM * sizeof(float),
                 hipMemcpyDeviceToDevice, stream);
  gemm(ctx, out_w, out_b, hfinal, Nn * T_STEPS, RDIM, RDIM, 1);  // h_final = hs + att

  // --- heads ---
  float* node_out = (float*)d_out;
  float* edge_out = (float*)d_out + (long)Nn * T_STEPS * 2;
  node_head_kernel<<<dim3((unsigned)cdiv((long)Nn * T_STEPS, 256)), 256, 0, stream>>>(
      hfinal, nd_w, nd_b, X_seq, node_out, Nn);
  edge_prep_kernel<<<dim3(1), 128, 0, stream>>>(ed_w, wAB);
  edge_head_kernel<<<dim3((unsigned)cdiv((long)E * T_STEPS, 8)), 256, 0, stream>>>(
      hfinal, wAB, ed_b, src, dst, edge_out, (long)E * T_STEPS);
}
